// ImprovedDirectDetectionLoss_16973710754164
// MI455X (gfx1250) — compile-verified, matrix operand order checked by reading, not run on hardware
//
#include <hip/hip_runtime.h>
#include <hip/hip_bf16.h>
#include <math.h>

typedef __attribute__((ext_vector_type(16))) _Float16 v16h;
typedef __attribute__((ext_vector_type(8)))  float    v8f;

// Problem constants (from reference setup_inputs)
constexpr int BB  = 16;
constexpr int CC  = 64;
constexpr int HH  = 80;
constexpr int WW  = 80;
constexpr int NN  = 32;
constexpr int HWp = HH * WW;              // 6400
constexpr float IOU_THRESH  = 0.3f;
constexpr float FOCAL_ALPHA = 0.25f;
constexpr float TOTAL_POS   = (float)BB * CC * HH * WW; // 6553600

// ---------------------------------------------------------------------------
// GIoU for xyxy boxes (pred vs gt). Matches reference box_giou.
// ---------------------------------------------------------------------------
__device__ __forceinline__ float giou_xyxy(const float4 p, const float4 g) {
    float a1 = (p.z - p.x) * (p.w - p.y);
    float a2 = (g.z - g.x) * (g.w - g.y);
    float ltx = fmaxf(p.x, g.x), lty = fmaxf(p.y, g.y);
    float rbx = fminf(p.z, g.z), rby = fminf(p.w, g.w);
    float iw = fmaxf(rbx - ltx, 0.0f), ih = fmaxf(rby - lty, 0.0f);
    float inter = iw * ih;
    float uni = a1 + a2 - inter;
    float iou = inter / uni;
    float ex = fminf(p.x, g.x), ey = fminf(p.y, g.y);
    float fx = fmaxf(p.z, g.z), fy = fmaxf(p.w, g.w);
    float ew = fmaxf(fx - ex, 0.0f), eh = fmaxf(fy - ey, 0.0f);
    float ae = ew * eh;
    return iou - (ae - uni) / ae;
}

// ---------------------------------------------------------------------------
// Kernel 0: zero the 4 global accumulators (l1, giou, focal, num_pos)
// ---------------------------------------------------------------------------
__global__ void k_init(float* acc) {
    if (threadIdx.x < 8) acc[threadIdx.x] = 0.0f;
}

// ---------------------------------------------------------------------------
// Kernel 1: per-(b,g) GIoU argmax over the label's H*W plane.
// One block per (b,g). meta[b*N+g] = {valid, mi, mj, 0}.
// Tie-break = lowest flat index (matches jnp.argmax first-occurrence).
// ---------------------------------------------------------------------------
__global__ void k_assign(const float4* __restrict__ pred4,   // [B,C,H,W] of float4
                         const float4* __restrict__ gtb4,    // [B,N] of float4
                         const int*    __restrict__ gtl,     // [B,N]
                         int4*         __restrict__ meta) {
    const int blk = blockIdx.x;
    const int b = blk / NN, g = blk % NN;
    const int lab = gtl[b * NN + g];
    const float4 gbox = gtb4[b * NN + g];
    const float4* plane = pred4 + (size_t)(b * CC + lab) * HWp;

    const int tid = threadIdx.x;
    float bestv = -INFINITY;
    int   besti = 0x7FFFFFFF;
    for (int p = tid; p < HWp; p += 256) {
        if (p + 256 < HWp) __builtin_prefetch(plane + p + 256, 0, 3);
        float v = giou_xyxy(plane[p], gbox);
        if (v > bestv) { bestv = v; besti = p; }   // ascending p => first max kept
    }

    __shared__ float sv[256];
    __shared__ int   si[256];
    sv[tid] = bestv; si[tid] = besti;
    __syncthreads();
    for (int s = 128; s > 0; s >>= 1) {
        if (tid < s) {
            float ov = sv[tid + s]; int oi = si[tid + s];
            if (ov > sv[tid] || (ov == sv[tid] && oi < si[tid])) { sv[tid] = ov; si[tid] = oi; }
        }
        __syncthreads();
    }
    if (tid == 0) {
        int idx = si[0];
        int valid = (sv[0] > IOU_THRESH) ? 1 : 0;
        meta[b * NN + g] = make_int4(valid, idx / WW, idx % WW, 0);
    }
}

// ---------------------------------------------------------------------------
// Kernel 2: main streaming pass. One block per (b,c) plane (1024 blocks,
// 256 threads, 25 iterations, float4 b128 loads). Computes positive mask /
// matched-gt on the fly from the compacted meta list, accumulates
// l1-sum, (1-giou)-sum, focal-sum (f32 LDS tree) and num_pos (exact
// matrix-engine reduction: v_wmma_f32_16x16x32_f16 with A = ones).
// ---------------------------------------------------------------------------
__global__ void k_main(const float4* __restrict__ pred4,
                       const float*  __restrict__ conf,
                       const float4* __restrict__ gtb4,
                       const int*    __restrict__ gtl,
                       const int4*   __restrict__ meta,
                       float*        __restrict__ acc) {
    const int bc = blockIdx.x;
    const int b = bc / CC, c = bc % CC;
    const int tid = threadIdx.x;

    __shared__ int    scnt;
    __shared__ int    smi[NN], smj[NN], sg[NN];
    __shared__ float4 sbox[NN];

    // Compact the GTs that hit this (b,c) plane, ascending g (max-g = last).
    if (tid == 0) {
        int m = 0;
        for (int g = 0; g < NN; ++g) {
            if (gtl[b * NN + g] == c) {
                int4 mt = meta[b * NN + g];
                if (mt.x) {
                    smi[m] = mt.y; smj[m] = mt.z; sg[m] = g;
                    sbox[m] = gtb4[b * NN + g];
                    ++m;
                }
            }
        }
        scnt = m;
    }
    __syncthreads();
    const int cnt = scnt;

    const float4* plane = pred4 + (size_t)bc * HWp;
    const float*  cpl   = conf  + (size_t)bc * HWp;

    float s_l1 = 0.0f, s_gi = 0.0f, s_fo = 0.0f;
    int   my_cnt = 0;

    #pragma unroll 1
    for (int it = 0; it < HWp / 256; ++it) {
        const int p = it * 256 + tid;
        if (p + 256 < HWp) __builtin_prefetch(plane + p + 256, 0, 3);
        const float4 pb = plane[p];
        const float  cv = cpl[p];
        const int i = p / WW, j = p - (p / WW) * WW;

        int matched = -1;
        for (int m = 0; m < cnt; ++m) {      // uniform trip count across block
            bool cov = (i >= smi[m] - 2) & (i <= smi[m] + 1) &
                       (j >= smj[m] - 2) & (j <= smj[m] + 1);
            if (cov) matched = m;            // ascending g: last wins == max g
        }
        const float t = (matched >= 0) ? 1.0f : 0.0f;

        if (matched >= 0) {
            const float4 gb = sbox[matched];
            float l1 = 0.25f * (fabsf(pb.x - gb.x) + fabsf(pb.y - gb.y) +
                                fabsf(pb.z - gb.z) + fabsf(pb.w - gb.w));
            s_l1 += l1;
            s_gi += 1.0f - giou_xyxy(pb, gb);
            ++my_cnt;
        }

        // focal loss (all positions)
        float pc = fminf(fmaxf(cv, 1e-6f), 1.0f - 1e-6f);
        float ce = -(t * logf(pc) + (1.0f - t) * log1pf(-pc));
        float pt = pc * t + (1.0f - pc) * (1.0f - t);
        float at = FOCAL_ALPHA * t + (1.0f - FOCAL_ALPHA) * (1.0f - t);
        float om = 1.0f - pt;
        s_fo += at * om * om * ce;
    }

    // ---- exact num_pos reduction on the matrix engine (wave32 WMMA) -------
    // Each lane's count (<=25, exact in f16) goes into one B slot; A = ones.
    // D = A*B + 0 gives f32 column sums; total = sum of c[0] over lanes 0..15.
    v16h a, bv = {};
    #pragma unroll
    for (int k = 0; k < 16; ++k) a[k] = (_Float16)1.0f;
    bv[0] = (_Float16)(float)my_cnt;
    v8f cz = {};
    v8f cd = __builtin_amdgcn_wmma_f32_16x16x32_f16(
        /*neg_a=*/false, a, /*neg_b=*/false, bv,
        /*c_mod=*/(short)0, cz, /*reuse_a=*/false, /*reuse_b=*/false);

    __shared__ float red0[256], red1[256], red2[256];
    __shared__ float wcnt[8 * 16];
    const int lane = tid & 31, wave = tid >> 5;
    if (lane < 16) wcnt[wave * 16 + lane] = cd[0];   // column sums (lanes 0-15)

    red0[tid] = s_l1; red1[tid] = s_gi; red2[tid] = s_fo;
    __syncthreads();
    for (int s = 128; s > 0; s >>= 1) {
        if (tid < s) {
            red0[tid] += red0[tid + s];
            red1[tid] += red1[tid + s];
            red2[tid] += red2[tid + s];
        }
        __syncthreads();
    }
    if (tid == 0) {
        float np = 0.0f;
        for (int k = 0; k < 8 * 16; ++k) np += wcnt[k];
        atomicAdd(acc + 0, red0[0]);
        atomicAdd(acc + 1, red1[0]);
        atomicAdd(acc + 2, red2[0]);
        atomicAdd(acc + 3, np);
    }
}

// ---------------------------------------------------------------------------
// Kernel 3: finalize 5 scalars
// ---------------------------------------------------------------------------
__global__ void k_final(const float* __restrict__ acc, float* __restrict__ out) {
    if (threadIdx.x == 0) {
        float np = acc[3];
        float denom = fmaxf(np, 1.0f);
        float l1 = acc[0] / denom;
        float gi = acc[1] / denom;
        float fo = acc[2] / TOTAL_POS;
        out[0] = l1;
        out[1] = gi;
        out[2] = fo;
        out[3] = l1 + 2.0f * gi + fo;         // lambdas 1,2,1
        out[4] = np / TOTAL_POS;
    }
}

// ---------------------------------------------------------------------------
extern "C" void kernel_launch(void* const* d_in, const int* in_sizes, int n_in,
                              void* d_out, int out_size, void* d_ws, size_t ws_size,
                              hipStream_t stream) {
    const float4* pred4 = (const float4*)d_in[0];      // [B,C,H,W,4] f32
    const float*  conf  = (const float*) d_in[1];      // [B,C,H,W]   f32
    // d_in[2] = cam (unused by the reference loss)
    const float4* gtb4  = (const float4*)d_in[3];      // [B,N,4]     f32
    const int*    gtl   = (const int*)   d_in[4];      // [B,N]       i32
    float* out = (float*)d_out;

    // workspace layout: [0,32)B accumulators ; [256, 256+512*16)B meta
    float* acc  = (float*)d_ws;
    int4*  meta = (int4*)((char*)d_ws + 256);

    k_init  <<<1, 32, 0, stream>>>(acc);
    k_assign<<<BB * NN, 256, 0, stream>>>(pred4, gtb4, gtl, meta);
    k_main  <<<BB * CC, 256, 0, stream>>>(pred4, conf, gtb4, gtl, meta, acc);
    k_final <<<1, 32, 0, stream>>>(acc, out);
}
